// LSTM_Model_12876311953630
// MI455X (gfx1250) — matrix operand order repaired and checked by reading, not compiled, not run on hardware
//
#include <hip/hip_runtime.h>
#include <hip/hip_bf16.h>

// ---------------------------------------------------------------------------
// MI455X (gfx1250) implementation.
// Precision plan: weights + recurrent state h in f16, WMMA f32 accumulation,
// cell state c in f32 registers.  All matrix math via v_wmma_f32_16x16x32_f16.
// beta (softmax over a size-1 axis) == 1 exactly, so Y3 == Y2 and the
// bW/bb/bv branch is mathematically dead — skipped.
// ---------------------------------------------------------------------------

typedef _Float16 f16;
typedef _Float16 v16h __attribute__((ext_vector_type(16)));
typedef _Float16 v8h  __attribute__((ext_vector_type(8)));
typedef float    v8f  __attribute__((ext_vector_type(8)));

union AB { v16h v; v8h h[2]; };

__device__ __forceinline__ v8f wmma_f16(v16h a, v16h b, v8f c) {
  // D = A(16x32 f16) x B(32x16 f16) + C(16x16 f32)
  return __builtin_amdgcn_wmma_f32_16x16x32_f16(
      /*neg_a=*/false, a, /*neg_b=*/false, b,
      /*c_mod=*/(short)0, c, /*reuse_a=*/false, /*reuse_b=*/false);
}

// gfx1250 has V_TANH_F32 (TRANS unit).  Use it for tanh AND sigmoid so the
// gate math is single-trans-op instead of exp + precise-division chains.
__device__ __forceinline__ float fast_tanh(float x) {
#if __has_builtin(__builtin_amdgcn_tanhf)
  return __builtin_amdgcn_tanhf(x);
#else
  return tanhf(x);
#endif
}
__device__ __forceinline__ float sigm(float x) {
  return 0.5f * fast_tanh(0.5f * x) + 0.5f;
}

__device__ __forceinline__ AB ldfragA(const f16* p) {  // row-major A chunk pair
  AB a;
  a.h[0] = *(const v8h*)(p);
  a.h[1] = *(const v8h*)(p + 16);
  return a;
}
__device__ __forceinline__ AB ldfragB(const f16* p) {  // contiguous 16 halves
  AB b;
  b.h[0] = *(const v8h*)(p);
  b.h[1] = *(const v8h*)(p + 8);
  return b;
}

// ---------------------------------------------------------------------------
// Weight swizzle: row-major f32 (K x N) -> per-fragment f16 layout for the
// WMMA B operand.  Fragment (slab, nt) is 32 lanes x 16 halves contiguous:
//   lane = (n&15) + ((kin>>3)&1)*16,  half = (kin&7) + (kin>>4)*8,  kin = k&31.
// ---------------------------------------------------------------------------
__global__ void swz_kernel(const float* __restrict__ src, f16* __restrict__ dst,
                           int K, int N) {
  int id = blockIdx.x * 256 + threadIdx.x;
  if (id >= K * N) return;
  int k = id / N, n = id % N;
  int NT   = N >> 4;
  int slab = k >> 5, kin = k & 31;
  int ln   = (n & 15) + ((kin & 8) << 1);
  int half = (kin & 7) + ((kin & 16) >> 1);
  int nt   = n >> 4;
  dst[(((size_t)slab * NT + nt) * 32 + ln) * 16 + half] = (f16)src[id];
}

// ---------------------------------------------------------------------------
// Stage 1: 42 branch-LSTMs per batch element.  One block per b (128 blocks),
// M = 48 rows (42 real + 6 zero pad), per step: z = h @ U1 via 48 WMMA tiles
// (each wave: 3 mt x 2 nt with dual interleaved accumulators), gates in VALU
// with c in registers, in-LDS group-mean -> res (no atomics).
// LDS: U1 fragments 32KB + z 24KB + h 6KB = 62KB.
// ---------------------------------------------------------------------------
__global__ __launch_bounds__(256) void lstm1_kernel(
    const float* __restrict__ y, const float* __restrict__ xp,
    const float* __restrict__ xn, const float* __restrict__ xi,
    const float* __restrict__ W1, const float* __restrict__ U1,
    const float* __restrict__ b1, float* __restrict__ res) {
  constexpr int T = 256, NB = 42, MR = 48, ZN = 256;
  __shared__ f16 U1s[2 * 16 * 32 * 16];  // 2 k-slabs x 16 n-tiles, swizzled
  __shared__ f16 z16[MR * ZN];
  __shared__ f16 h16[MR * 64];
  const int tid = threadIdx.x, lane = tid & 31, wave = tid >> 5;
  const int b = blockIdx.x;

  // One-time: swizzle U1 (64 x 256) into LDS as f16 B-fragments.
  for (int idx = tid; idx < 16384; idx += 256) {
    int i = idx & 15, ln = (idx >> 4) & 31, nt = (idx >> 9) & 15, s = idx >> 13;
    int k = s * 32 + ((ln & 16) >> 1) + (i & 7) + ((i & 8) << 1);
    int n = nt * 16 + (ln & 15);
    U1s[idx] = (f16)U1[k * 256 + n];
  }
  for (int idx = tid; idx < MR * 64; idx += 256) h16[idx] = (f16)0.f;

  // Per-thread gate-phase constants: j fixed, up to 11 (m,j) cells.
  const int j = tid & 63;
  float w1g[4], b1g[4];
#pragma unroll
  for (int g = 0; g < 4; ++g) { w1g[g] = W1[g * 64 + j]; b1g[g] = b1[g * 64 + j]; }
  const float* xsrc[11];
  float c[11];
#pragma unroll
  for (int q = 0; q < 11; ++q) {
    c[q] = 0.f;
    int m = (tid >> 6) + 4 * q;
    const float* p = y;
    if (m < NB) {
      if (m == 0)       p = y  + (size_t)b * T;
      else if (m <= 16) p = xp + (size_t)(b * 16 + (m - 1 )) * T;
      else if (m <= 32) p = xn + (size_t)(b * 16 + (m - 17)) * T;
      else              p = xi + (size_t)(b * 9  + (m - 33)) * T;
    }
    xsrc[q] = p;
  }
  const int mA = lane & 15, offA = (lane & 16) ? 8 : 0;
  const int nt0 = wave, nt1 = wave + 8;
  __syncthreads();

  for (int t = 0; t < T; ++t) {
    // ---- z = h @ U1 (WMMA): per wave 3 mt x {nt0, nt1}, dual accumulators --
#pragma unroll
    for (int mt = 0; mt < 3; ++mt) {
      const f16* arow = &h16[(mt * 16 + mA) * 64 + offA];
      AB a0 = ldfragA(arow);        // K 0..31
      AB a1 = ldfragA(arow + 32);   // K 32..63
      AB b00 = ldfragB(&U1s[((0 * 16 + nt0) * 32 + lane) * 16]);
      AB b10 = ldfragB(&U1s[((0 * 16 + nt1) * 32 + lane) * 16]);
      AB b01 = ldfragB(&U1s[((1 * 16 + nt0) * 32 + lane) * 16]);
      AB b11 = ldfragB(&U1s[((1 * 16 + nt1) * 32 + lane) * 16]);
      v8f acc0 = {}, acc1 = {};
      acc0 = wmma_f16(a0.v, b00.v, acc0);
      acc1 = wmma_f16(a0.v, b10.v, acc1);   // independent: fills hazard slots
      acc0 = wmma_f16(a1.v, b01.v, acc0);
      acc1 = wmma_f16(a1.v, b11.v, acc1);
      int mbase = mt * 16 + ((lane & 16) >> 1);
      int n0 = nt0 * 16 + (lane & 15);
#pragma unroll
      for (int r = 0; r < 8; ++r) {
        z16[(mbase + r) * ZN + n0]       = (f16)acc0[r];
        z16[(mbase + r) * ZN + n0 + 128] = (f16)acc1[r];
      }
    }
    __syncthreads();

    // ---- gates (branch uniform per wave: tid>>6 constant in a wave) ----
#pragma unroll
    for (int q = 0; q < 11; ++q) {
      int m = (tid >> 6) + 4 * q;
      if (m < NB) {
        float xv = xsrc[q][t];
        const f16* zr = &z16[m * ZN];
        float zi = (float)zr[      j] + xv * w1g[0] + b1g[0];
        float zf = (float)zr[ 64 + j] + xv * w1g[1] + b1g[1];
        float zg = (float)zr[128 + j] + xv * w1g[2] + b1g[2];
        float zo = (float)zr[192 + j] + xv * w1g[3] + b1g[3];
        float c2 = sigm(zf) * c[q] + sigm(zi) * fast_tanh(zg);
        c[q] = c2;
        h16[m * 64 + j] = (f16)(sigm(zo) * fast_tanh(c2));
      }
    }
    __syncthreads();

    // ---- group means -> res[b, t, 4*64] (no global atomics) ----
    {
      int seg = tid >> 6;
      int r0 = 0, r1 = 1; float scl = 1.f;
      if (seg == 1)      { r0 = 1;  r1 = 17; scl = 1.f / 16.f; }
      else if (seg == 2) { r0 = 17; r1 = 33; scl = 1.f / 16.f; }
      else if (seg == 3) { r0 = 33; r1 = 42; scl = 1.f / 9.f;  }
      float sum = 0.f;
      for (int m = r0; m < r1; ++m) sum += (float)h16[m * 64 + j];
      res[((size_t)(b * T + t)) * 256 + seg * 64 + j] = sum * scl;
    }
    __syncthreads();
  }
}

// ---------------------------------------------------------------------------
// Stage 2: MI-LSTM.  8 blocks x 16 batch rows.  Weights streamed from L2 as
// pre-swizzled f16 fragments.  Pass1: 40 WMMA tiles (z + 3 g_extra GEMMs
// sharing the h half of A).  Pass2: l_i @ Aw with LDS-atomic reduce vs c.
// h2 is scattered directly into stage-3's A-fragment layout (AF).
// ---------------------------------------------------------------------------
__global__ __launch_bounds__(256) void milstm_kernel(
    const float* __restrict__ res, const f16* __restrict__ WmF,
    const f16* __restrict__ WgF, const f16* __restrict__ AwF,
    const float* __restrict__ bm, const float* __restrict__ bg,
    const float* __restrict__ ab, f16* __restrict__ AF) {
  constexpr int T = 256;
  __shared__ f16 z16[16 * 640];
  __shared__ f16 x16[16 * 256];
  __shared__ f16 h16[16 * 64];
  __shared__ f16 l16[4 * 16 * 64];
  __shared__ float c32[16 * 64];
  __shared__ float usum[4 * 16];
  __shared__ float bias[640];
  const int tid = threadIdx.x, lane = tid & 31, wave = tid >> 5;
  const int b0 = blockIdx.x * 16;
  const int mA = lane & 15, offA = (lane & 16) ? 8 : 0;

  for (int e = tid; e < 640; e += 256) bias[e] = (e < 448) ? bm[e] : bg[e - 448];
  for (int e = tid; e < 16 * 64; e += 256) { c32[e] = 0.f; h16[e] = (f16)0.f; }
  float ab4[4];
#pragma unroll
  for (int i2 = 0; i2 < 4; ++i2) ab4[i2] = ab[i2];
  float creg[4] = {0.f, 0.f, 0.f, 0.f};
  __syncthreads();

  for (int t = 0; t < T; ++t) {
    // load x0..x3 for this step; prefetch next step's tile into cache
#pragma unroll
    for (int q = 0; q < 16; ++q) {
      int e = tid + 256 * q;
      int m = e >> 8, col = e & 255;
      const float* src = &res[((size_t)((b0 + m) * T + t)) * 256 + col];
      x16[e] = (f16)src[0];
      if (t + 1 < T) __builtin_prefetch(src + 256, 0, 1);  // global_prefetch_b8
    }
    __syncthreads();

    // pass 1: z (28 tiles, Wm) + g_extra (12 tiles, Wg_k)
#pragma unroll
    for (int u = 0; u < 5; ++u) {
      int tile = wave + 8 * u;
      int asel, nt, ncol, NTb; const f16* bmat;
      if (tile < 28) { asel = 0; nt = tile; bmat = WmF; NTb = 28; ncol = tile * 16; }
      else {
        int k = (tile - 28) >> 2; nt = (tile - 28) & 3;
        asel = k + 1; bmat = WgF + (size_t)k * 16 * 512; NTb = 4;
        ncol = 448 + (tile - 28) * 16;
      }
      v8f acc = {};
#pragma unroll
      for (int s = 0; s < 4; ++s) {            // K = 128 -> 4 slabs
        const f16* ap = (s < 2)
            ? &x16[mA * 256 + asel * 64 + s * 32 + offA]
            : &h16[mA * 64 + (s - 2) * 32 + offA];
        AB a  = ldfragA(ap);
        AB bf = ldfragB(bmat + ((size_t)(s * NTb + nt) * 32 + lane) * 16);
        acc = wmma_f16(a.v, bf.v, acc);
      }
      int mbase = (lane & 16) >> 1, n = ncol + (lane & 15);
#pragma unroll
      for (int r = 0; r < 8; ++r) z16[(mbase + r) * 640 + n] = (f16)acc[r];
    }
    __syncthreads();

    // l = sigmoid(s_i) * new_in_i
#pragma unroll
    for (int q = 0; q < 16; ++q) {
      int e = tid + 256 * q;
      int i2 = e >> 10, m = (e >> 6) & 15, jj = e & 63;
      float sv = (float)z16[m * 640 + 192 + i2 * 64 + jj] + bias[192 + i2 * 64 + jj];
      float gin;
      if (i2 == 0) gin = (float)z16[m * 640 + 128 + jj] + bias[128 + jj];
      else         gin = (float)z16[m * 640 + 448 + (i2 - 1) * 64 + jj]
                        + bias[448 + (i2 - 1) * 64 + jj];
      l16[(i2 * 16 + m) * 64 + jj] = (f16)(sigm(sv) * fast_tanh(gin));
    }
    if (tid < 64) usum[tid] = 0.f;
    __syncthreads();

    // pass 2: u_partial[i,m] = sum_n (l_i @ Aw)[m,n] * c[m,n]
#pragma unroll
    for (int u = 0; u < 2; ++u) {
      int p = wave + 8 * u;
      int i2 = p >> 2, nt = p & 3;
      v8f acc = {};
#pragma unroll
      for (int s = 0; s < 2; ++s) {
        AB a  = ldfragA(&l16[(i2 * 16 + mA) * 64 + s * 32 + offA]);
        AB bf = ldfragB(AwF + ((size_t)(s * 4 + nt) * 32 + lane) * 16);
        acc = wmma_f16(a.v, bf.v, acc);
      }
      int mbase = (lane & 16) >> 1, n = nt * 16 + (lane & 15);
#pragma unroll
      for (int r = 0; r < 8; ++r)
        atomicAdd(&usum[i2 * 16 + mbase + r], acc[r] * c32[(mbase + r) * 64 + n]);
    }
    __syncthreads();

    // softmax over i, L, cell update, h2 -> h16 + AF (stage-3 A fragments)
#pragma unroll
    for (int q = 0; q < 4; ++q) {
      int e = tid + 256 * q;
      int m = e >> 6, jj = e & 63;
      float u0 = fast_tanh(usum[ 0 + m] + ab4[0]);
      float u1 = fast_tanh(usum[16 + m] + ab4[1]);
      float u2 = fast_tanh(usum[32 + m] + ab4[2]);
      float u3 = fast_tanh(usum[48 + m] + ab4[3]);
      float mx = fmaxf(fmaxf(u0, u1), fmaxf(u2, u3));
      float e0 = __expf(u0 - mx), e1 = __expf(u1 - mx);
      float e2 = __expf(u2 - mx), e3 = __expf(u3 - mx);
      float inv = 1.f / (e0 + e1 + e2 + e3);
      float L = (e0 * (float)l16[( 0 + m) * 64 + jj] +
                 e1 * (float)l16[(16 + m) * 64 + jj] +
                 e2 * (float)l16[(32 + m) * 64 + jj] +
                 e3 * (float)l16[(48 + m) * 64 + jj]) * inv;
      float fpre = (float)z16[m * 640 + jj]      + bias[jj];
      float opre = (float)z16[m * 640 + 64 + jj] + bias[64 + jj];
      float c2 = creg[q] * sigm(fpre + 1.f) + L;
      creg[q] = c2;
      float h2 = fast_tanh(c2) * sigm(opre);
      c32[m * 64 + jj] = c2;
      h16[m * 64 + jj] = (f16)h2;
      int col = t * 64 + jj;
      int slab = col >> 5, kin = col & 31, grp = kin >> 3;
      int ln = m + ((grp & 1) << 4);
      int hf = (kin & 7) + ((grp >> 1) << 3);
      AF[(((size_t)blockIdx.x * 512 + slab) * 32 + ln) * 16 + hf] = (f16)h2;
    }
    __syncthreads();
  }
}

// ---------------------------------------------------------------------------
// Stage 3a: r0 = relu(Y3 @ Dw0 + Db0), K = 16384.  Even/odd slab pairs feed
// two independent accumulators to hide WMMA->WMMA RAW hazards.
// ---------------------------------------------------------------------------
__global__ __launch_bounds__(128) void mlp0_kernel(
    const f16* __restrict__ AF, const f16* __restrict__ Dw0F,
    const float* __restrict__ Db0, float* __restrict__ r0) {
  const int tid = threadIdx.x, lane = tid & 31, nt = tid >> 5;
  v8f acc0 = {}, acc1 = {};
  for (int slab = 0; slab < 512; slab += 2) {
    AB a0 = ldfragB(AF + (((size_t)blockIdx.x * 512 + slab) * 32 + lane) * 16);
    AB b0 = ldfragB(Dw0F + (((size_t)slab * 4 + nt) * 32 + lane) * 16);
    AB a1 = ldfragB(AF + (((size_t)blockIdx.x * 512 + slab + 1) * 32 + lane) * 16);
    AB b1 = ldfragB(Dw0F + (((size_t)(slab + 1) * 4 + nt) * 32 + lane) * 16);
    acc0 = wmma_f16(a0.v, b0.v, acc0);
    acc1 = wmma_f16(a1.v, b1.v, acc1);
  }
  v8f acc = acc0 + acc1;
  int mbase = blockIdx.x * 16 + ((lane & 16) >> 1);
  int n = nt * 16 + (lane & 15);
  float bv = Db0[n];
#pragma unroll
  for (int r = 0; r < 8; ++r)
    r0[(size_t)(mbase + r) * 64 + n] = fmaxf(acc[r] + bv, 0.f);
}

// ---------------------------------------------------------------------------
// Stage 3b: three 64x64 layers (WMMA, LDS ping-pong) + final 64-dot.
// ---------------------------------------------------------------------------
__global__ __launch_bounds__(256) void mlp_tail_kernel(
    const float* __restrict__ r0,
    const f16* __restrict__ Dw1F, const float* __restrict__ Db1,
    const f16* __restrict__ Dw2F, const float* __restrict__ Db2,
    const f16* __restrict__ Dw3F, const float* __restrict__ Db3,
    const float* __restrict__ Dw4, const float* __restrict__ Db4,
    float* __restrict__ out) {
  __shared__ f16 buf[2][128 * 64];
  const int tid = threadIdx.x, lane = tid & 31, wave = tid >> 5;
  const int mA = lane & 15, offA = (lane & 16) ? 8 : 0;
  for (int e = tid; e < 128 * 64; e += 256) buf[0][e] = (f16)r0[e];
  __syncthreads();
  const f16* Wf[3] = {Dw1F, Dw2F, Dw3F};
  const float* Bb[3] = {Db1, Db2, Db3};
#pragma unroll
  for (int lyr = 0; lyr < 3; ++lyr) {
    const f16* src = buf[lyr & 1];
    f16* dst = buf[(lyr + 1) & 1];
#pragma unroll
    for (int u = 0; u < 4; ++u) {
      int p = wave + 8 * u;
      int mt = p >> 2, nt = p & 3;
      v8f acc = {};
#pragma unroll
      for (int s = 0; s < 2; ++s) {
        AB a  = ldfragA(&src[(mt * 16 + mA) * 64 + s * 32 + offA]);
        AB bf = ldfragB(Wf[lyr] + ((size_t)(s * 4 + nt) * 32 + lane) * 16);
        acc = wmma_f16(a.v, bf.v, acc);
      }
      int mbase = mt * 16 + ((lane & 16) >> 1);
      int n = nt * 16 + (lane & 15);
      float bv = Bb[lyr][n];
#pragma unroll
      for (int r = 0; r < 8; ++r)
        dst[(mbase + r) * 64 + n] = (f16)fmaxf(acc[r] + bv, 0.f);
    }
    __syncthreads();
  }
  if (tid < 128) {
    float s = Db4[0];
    const f16* row = &buf[1][tid * 64];
    for (int j2 = 0; j2 < 64; ++j2) s += (float)row[j2] * Dw4[j2];
    out[tid] = s;
  }
}

// ---------------------------------------------------------------------------
extern "C" void kernel_launch(void* const* d_in, const int* in_sizes, int n_in,
                              void* d_out, int out_size, void* d_ws, size_t ws_size,
                              hipStream_t stream) {
  const float* y   = (const float*)d_in[0];
  const float* xp  = (const float*)d_in[1];
  const float* xn  = (const float*)d_in[2];
  const float* xi  = (const float*)d_in[3];
  const float* W1  = (const float*)d_in[4];
  const float* U1  = (const float*)d_in[5];
  const float* b1  = (const float*)d_in[6];
  const float* Wm  = (const float*)d_in[7];
  const float* bm  = (const float*)d_in[8];
  const float* Wg  = (const float*)d_in[9];
  const float* bg  = (const float*)d_in[10];
  const float* Aw  = (const float*)d_in[11];
  const float* ab  = (const float*)d_in[12];
  // d_in[13..15] = bW, bb, bv: provably unused (softmax over singleton axis == 1)
  const float* Dw0 = (const float*)d_in[16];
  const float* Db0 = (const float*)d_in[17];
  const float* Dw1 = (const float*)d_in[18];
  const float* Db1 = (const float*)d_in[19];
  const float* Dw2 = (const float*)d_in[20];
  const float* Db2 = (const float*)d_in[21];
  const float* Dw3 = (const float*)d_in[22];
  const float* Db3 = (const float*)d_in[23];
  const float* Dw4 = (const float*)d_in[24];
  const float* Db4 = (const float*)d_in[25];

  char* w = (char*)d_ws;
  float* res = (float*)w; w += (size_t)128 * 256 * 256 * 4;  // 32 MB
  f16* WmF   = (f16*)w;   w += (size_t)112 * 512 * 2;
  f16* WgF   = (f16*)w;   w += (size_t)48 * 512 * 2;
  f16* AwF   = (f16*)w;   w += (size_t)8 * 512 * 2;
  f16* Dw0F  = (f16*)w;   w += (size_t)2048 * 512 * 2;       // 2 MB
  f16* Dw1F  = (f16*)w;   w += (size_t)8 * 512 * 2;
  f16* Dw2F  = (f16*)w;   w += (size_t)8 * 512 * 2;
  f16* Dw3F  = (f16*)w;   w += (size_t)8 * 512 * 2;
  f16* AF    = (f16*)w;   w += (size_t)8 * 512 * 512 * 2;    // 4 MB
  float* r0  = (float*)w; w += (size_t)128 * 64 * 4;

  auto nblk = [](int n) { return (n + 255) / 256; };
  swz_kernel<<<nblk(128 * 448), 256, 0, stream>>>(Wm, WmF, 128, 448);
  for (int k = 0; k < 3; ++k)
    swz_kernel<<<nblk(128 * 64), 256, 0, stream>>>(
        Wg + (size_t)k * 128 * 64, WgF + (size_t)k * 16 * 512, 128, 64);
  swz_kernel<<<nblk(64 * 64), 256, 0, stream>>>(Aw, AwF, 64, 64);
  swz_kernel<<<nblk(16384 * 64), 256, 0, stream>>>(Dw0, Dw0F, 16384, 64);
  swz_kernel<<<nblk(64 * 64), 256, 0, stream>>>(Dw1, Dw1F, 64, 64);
  swz_kernel<<<nblk(64 * 64), 256, 0, stream>>>(Dw2, Dw2F, 64, 64);
  swz_kernel<<<nblk(64 * 64), 256, 0, stream>>>(Dw3, Dw3F, 64, 64);

  lstm1_kernel<<<128, 256, 0, stream>>>(y, xp, xn, xi, W1, U1, b1, res);
  milstm_kernel<<<8, 256, 0, stream>>>(res, WmF, WgF, AwF, bm, bg, ab, AF);
  mlp0_kernel<<<8, 128, 0, stream>>>(AF, Dw0F, Db0, r0);
  mlp_tail_kernel<<<1, 256, 0, stream>>>(r0, Dw1F, Db1, Dw2F, Db2, Dw3F, Db3,
                                         Dw4, Db4, (float*)d_out);
  (void)in_sizes; (void)n_in; (void)out_size; (void)ws_size;
}
